// CausalSelfAttention_6201932775854
// MI455X (gfx1250) — matrix operand order repaired and checked by reading, not compiled, not run on hardware
//
#include <hip/hip_runtime.h>
#include <hip/hip_bf16.h>
#include <cstdint>

typedef __attribute__((ext_vector_type(8)))  __bf16 bf16x8;
typedef __attribute__((ext_vector_type(16))) __bf16 bf16x16;
typedef __attribute__((ext_vector_type(8)))  float  f32x8;
typedef __attribute__((ext_vector_type(4)))  unsigned int u32x4;
typedef __attribute__((ext_vector_type(8)))  int i32x8;
typedef __attribute__((ext_vector_type(4)))  int i32x4;

constexpr int B  = 2;
constexpr int S  = 2048;
constexpr int E  = 1024;
constexpr int H  = 16;
constexpr int DH = 64;
constexpr int F3 = 3 * DH;   // 192

// ---- workspace layout (bytes) ----
constexpr size_t OFF_XB  = 0;            // x bf16        [B][S][E]      8 MiB
constexpr size_t OFF_WKT = 8ull  << 20;  // Wkqv^T bf16   [H][F3][E]     6 MiB
constexpr size_t OFF_WOT = 14ull << 20;  // Wo^T bf16     [E][E]         2 MiB
constexpr size_t OFF_K   = 16ull << 20;  // K bf16        [B*H][S][DH]   8 MiB
constexpr size_t OFF_Q   = 24ull << 20;  // Q bf16        [B*H][S][DH]   8 MiB
constexpr size_t OFF_VT  = 32ull << 20;  // V^T bf16      [B*H][DH][S]   8 MiB
constexpr size_t OFF_SA  = 40ull << 20;  // attn out bf16 [B][S][E]      8 MiB

__device__ __forceinline__ bf16x16 join16(bf16x8 lo, bf16x8 hi) {
  return __builtin_shufflevector(lo, hi, 0,1,2,3,4,5,6,7,8,9,10,11,12,13,14,15);
}

// A-matrix 16x32 bf16 fragment (ISA 7.12.2): lane = (m = lane&15, half = lane>>4)
// elements 0..7  = A[m][half*8 + 0..7], elements 8..15 = A[m][16 + half*8 + 0..7]
__device__ __forceinline__ bf16x16 load_a_frag(const __bf16* base, int lda, int lane) {
  const int m = lane & 15, hf = lane >> 4;
  const __bf16* p = base + (size_t)m * lda + hf * 8;
  bf16x8 lo = *(const bf16x8*)(p);
  bf16x8 hi = *(const bf16x8*)(p + 16);
  return join16(lo, hi);
}

// B-matrix 32x16 bf16 fragment, read from B^T (N-major, ldb = K stride):
// lane = (n = lane&15, half): elements 0..7 = B^T[n][half*8 + 0..7], 8..15 = B^T[n][16 + half*8 + 0..7]
__device__ __forceinline__ bf16x16 load_b_frag(const __bf16* baseT, int ldb, int lane) {
  const int n = lane & 15, hf = lane >> 4;
  const __bf16* p = baseT + (size_t)n * ldb + hf * 8;
  bf16x8 lo = *(const bf16x8*)(p);
  bf16x8 hi = *(const bf16x8*)(p + 16);
  return join16(lo, hi);
}

__device__ __forceinline__ f32x8 wmma_bf16(bf16x16 a, bf16x16 b, f32x8 c) {
  return __builtin_amdgcn_wmma_f32_16x16x32_bf16(false, a, false, b, (short)0, c,
                                                 false, false);
}

// ---- Tensor Data Mover: 2-D bf16 tile -> LDS (ISA 8.3/8.4 D# packing) ----
// dims/strides in elements (data_size = 2 bytes), addresses in bytes.
// This toolchain exposes the 6-arg builtin: (g0, g1, g2, g3, g4, cpol).
__device__ __forceinline__ void tdm_load_2d(uint32_t lds_off, const void* gptr,
                                            uint32_t tile_x, uint32_t tile_y,
                                            uint32_t tensor_x, uint32_t tensor_y,
                                            uint64_t stride_x) {
  const uint64_t ga = (uint64_t)(uintptr_t)gptr;
  u32x4 g0;
  g0[0] = 1u;                                   // count=1, user descriptor
  g0[1] = lds_off;                              // lds_addr[31:0]
  g0[2] = (uint32_t)ga;                         // global_addr[31:0]
  g0[3] = (uint32_t)((ga >> 32) & 0x01FFFFFFu)  // global_addr[56:32]
        | (2u << 30);                           // type = 2 ("image")
  i32x8 g1;
  g1[0] = (int)(1u << 16);                      // data_size=1 (2B), wg_mask=0
  g1[1] = (int)((tensor_x & 0xFFFFu) << 16);    // tensor_dim0[15:0]
  g1[2] = (int)(((tensor_x >> 16) & 0xFFFFu) | ((tensor_y & 0xFFFFu) << 16));
  g1[3] = (int)(((tensor_y >> 16) & 0xFFFFu) | ((tile_x & 0xFFFFu) << 16));
  g1[4] = (int)(tile_y & 0xFFFFu);              // tile_dim1 (tile_dim2 = 0)
  g1[5] = (int)(uint32_t)(stride_x & 0xFFFFFFFFu);        // dim0_stride[31:0]
  g1[6] = (int)(uint32_t)((stride_x >> 32) & 0xFFFFu);    // dim0_stride[47:32]
  g1[7] = 0;
  const i32x4 gz4 = {0, 0, 0, 0};               // 2-D: groups 2/3 unused
  const i32x8 gz8 = {0, 0, 0, 0, 0, 0, 0, 0};
  __builtin_amdgcn_tensor_load_to_lds(g0, g1, gz4, gz4, gz8, 0);
}

// ---------------- conversion kernels ----------------
__global__ void cvt_x_kernel(const float* __restrict__ x, __bf16* __restrict__ xb, int n) {
  for (int i = blockIdx.x * blockDim.x + threadIdx.x; i < n; i += gridDim.x * blockDim.x)
    xb[i] = (__bf16)x[i];
}

// Wkqv [H][E][F3] -> WkT [H][F3][E]
__global__ void cvt_wkqv_kernel(const float* __restrict__ w, __bf16* __restrict__ wt) {
  const int n = H * F3 * E;
  for (int i = blockIdx.x * blockDim.x + threadIdx.x; i < n; i += gridDim.x * blockDim.x) {
    int e = i % E;
    int t = i / E;
    int f = t % F3;
    int h = t / F3;
    wt[i] = (__bf16)w[((size_t)h * E + e) * F3 + f];
  }
}

// Wo [E][E] -> WoT [n][k] = Wo[k][n]
__global__ void cvt_wo_kernel(const float* __restrict__ w, __bf16* __restrict__ wt) {
  const int n = E * E;
  for (int i = blockIdx.x * blockDim.x + threadIdx.x; i < n; i += gridDim.x * blockDim.x) {
    int k = i % E;
    int c = i / E;
    wt[i] = (__bf16)w[(size_t)k * E + c];
  }
}

// ---------------- KQV projection ----------------
// grid: (S/256, F3/32, B*H), block 256 (8 waves); wave -> one 32x32 tile
__global__ void __launch_bounds__(256)
kqv_kernel(const __bf16* __restrict__ xb, const __bf16* __restrict__ wkT,
           const float* __restrict__ bkqv,
           __bf16* __restrict__ Km, __bf16* __restrict__ Qm, __bf16* __restrict__ Vt) {
  const int lane = threadIdx.x & 31;
  const int wave = threadIdx.x >> 5;
  const int bh = blockIdx.z;
  const int b = bh >> 4, h = bh & 15;
  const int row0 = blockIdx.x * 256 + wave * 32;    // sequence tile
  const int col0 = blockIdx.y * 32;                 // feature tile (0..191)

  const __bf16* A  = xb  + ((size_t)b * S + row0) * E;
  const __bf16* Bt = wkT + ((size_t)h * F3 + col0) * E;

  f32x8 c00 = {}, c01 = {}, c10 = {}, c11 = {};
#pragma unroll 4
  for (int k0 = 0; k0 < E; k0 += 32) {
    bf16x16 a0 = load_a_frag(A + k0, E, lane);
    bf16x16 a1 = load_a_frag(A + (size_t)16 * E + k0, E, lane);
    bf16x16 b0 = load_b_frag(Bt + k0, E, lane);
    bf16x16 b1 = load_b_frag(Bt + (size_t)16 * E + k0, E, lane);
    c00 = wmma_bf16(a0, b0, c00);
    c01 = wmma_bf16(a0, b1, c01);
    c10 = wmma_bf16(a1, b0, c10);
    c11 = wmma_bf16(a1, b1, c11);
  }

  const int n = lane & 15, hf = lane >> 4;
  const float* bias = bkqv + h * F3;
  f32x8 cs[4] = {c00, c01, c10, c11};
#pragma unroll
  for (int t = 0; t < 4; ++t) {
    const int f = col0 + (t & 1) * 16 + n;
    const int rbase = row0 + (t >> 1) * 16;
    const float bv = bias[f];
#pragma unroll
    for (int r = 0; r < 8; ++r) {
      const int s = rbase + r + 8 * hf;
      const __bf16 v = (__bf16)(cs[t][r] + bv);
      if (f < DH)            Km[((size_t)bh * S + s) * DH + f] = v;
      else if (f < 2 * DH)   Qm[((size_t)bh * S + s) * DH + (f - DH)] = v;
      else                   Vt[((size_t)bh * DH + (f - 2 * DH)) * S + s] = v;
    }
  }
}

// ---------------- flash attention (block-cooperative, TDM-staged K/V) ------
// grid: (S/128, 1, B*H), block 256 (8 waves); wave -> 16 query rows.
// Wave 0 DMAs the shared K (32xDH) and V^T (DHx32) chunk into LDS via the
// Tensor Data Mover; all waves consume it from LDS.
__global__ void __launch_bounds__(256)
attn_kernel(const __bf16* __restrict__ Qm, const __bf16* __restrict__ Km,
            const __bf16* __restrict__ Vt, __bf16* __restrict__ sa) {
  __shared__ __align__(16) __bf16 ldsK[32][DH];     // 4 KB
  __shared__ __align__(16) __bf16 ldsV[DH][32];     // 4 KB
  __shared__ __align__(16) __bf16 lds_p[8][16][32]; // 8 KB (per-wave P tiles)

  const int lane = threadIdx.x & 31;
  const int wave = threadIdx.x >> 5;
  const int bh = blockIdx.z;
  const int b = bh >> 4, h = bh & 15;
  const int q0 = blockIdx.x * 128 + wave * 16;

  const __bf16* Qb = Qm + ((size_t)bh * S + q0) * DH;
  const __bf16* Kb = Km + (size_t)bh * S * DH;
  const __bf16* Vb = Vt + (size_t)bh * DH * S;

  const int n = lane & 15, hf = lane >> 4;
  const float NEG_INF = -__builtin_inff();
  const float inv_sqrt_d = 0.125f;      // 1/sqrt(64)

  const bf16x16 qa0 = load_a_frag(Qb + 0, DH, lane);
  const bf16x16 qa1 = load_a_frag(Qb + 32, DH, lane);

  f32x8 o0 = {}, o1 = {}, o2 = {}, o3 = {};
  float mrow[8], lrow[8];
#pragma unroll
  for (int r = 0; r < 8; ++r) { mrow[r] = NEG_INF; lrow[r] = 0.f; }

  const uint32_t ldsK_off = (uint32_t)(uintptr_t)&ldsK[0][0];
  const uint32_t ldsV_off = (uint32_t)(uintptr_t)&ldsV[0][0];

  const int KV_END = (blockIdx.x + 1) * 128;     // block-uniform causal bound
  for (int kv0 = 0; kv0 < KV_END; kv0 += 32) {
    if (wave == 0) {
      // K rows kv0..kv0+31 (row-major [S][DH])
      tdm_load_2d(ldsK_off, Kb + (size_t)kv0 * DH,
                  /*tile_x=*/DH, /*tile_y=*/32,
                  /*tensor_x=*/DH, /*tensor_y=*/(uint32_t)(S - kv0),
                  /*stride_x=*/DH);
      // V^T cols kv0..kv0+31 of all DH rows (row-major [DH][S])
      tdm_load_2d(ldsV_off, Vb + kv0,
                  /*tile_x=*/32, /*tile_y=*/DH,
                  /*tensor_x=*/(uint32_t)(S - kv0), /*tensor_y=*/DH,
                  /*stride_x=*/S);
      __builtin_amdgcn_s_wait_tensorcnt(0);
    }
    __syncthreads();

    if (kv0 < q0 + 16) {                 // this wave's causal range
      // scores: two 16-col tiles, K-dim 64, B-fragments from LDS
      f32x8 s0 = {}, s1 = {};
      s0 = wmma_bf16(qa0, load_b_frag(&ldsK[0][0],  DH, lane), s0);
      s0 = wmma_bf16(qa1, load_b_frag(&ldsK[0][32], DH, lane), s0);
      s1 = wmma_bf16(qa0, load_b_frag(&ldsK[16][0],  DH, lane), s1);
      s1 = wmma_bf16(qa1, load_b_frag(&ldsK[16][32], DH, lane), s1);

#pragma unroll
      for (int r = 0; r < 8; ++r) {
        const int q = q0 + r + 8 * hf;
        float v0 = s0[r] * inv_sqrt_d;
        float v1 = s1[r] * inv_sqrt_d;
        if (kv0 + n > q)      v0 = NEG_INF;
        if (kv0 + 16 + n > q) v1 = NEG_INF;
        float mx = fmaxf(v0, v1);
#pragma unroll
        for (int off = 1; off < 16; off <<= 1) mx = fmaxf(mx, __shfl_xor(mx, off, 32));
        const float mnew = fmaxf(mrow[r], mx);
        const float scale = __expf(mrow[r] - mnew);
        const float p0 = __expf(v0 - mnew);
        const float p1 = __expf(v1 - mnew);
        float rs = p0 + p1;
#pragma unroll
        for (int off = 1; off < 16; off <<= 1) rs += __shfl_xor(rs, off, 32);
        lrow[r] = lrow[r] * scale + rs;
        mrow[r] = mnew;
        o0[r] *= scale; o1[r] *= scale; o2[r] *= scale; o3[r] *= scale;
        lds_p[wave][r + 8 * hf][n]      = (__bf16)p0;
        lds_p[wave][r + 8 * hf][16 + n] = (__bf16)p1;
      }

      // P(16x32) @ V(32x64): A from LDS P tile, B from staged V^T
      const bf16x16 pa = load_a_frag(&lds_p[wave][0][0], 32, lane);
      o0 = wmma_bf16(pa, load_b_frag(&ldsV[0][0],  32, lane), o0);
      o1 = wmma_bf16(pa, load_b_frag(&ldsV[16][0], 32, lane), o1);
      o2 = wmma_bf16(pa, load_b_frag(&ldsV[32][0], 32, lane), o2);
      o3 = wmma_bf16(pa, load_b_frag(&ldsV[48][0], 32, lane), o3);
    }
    __syncthreads();                     // protect staging buffers
  }

  // normalize and store head-concat bf16 [b][s][h*DH + d]
#pragma unroll
  for (int r = 0; r < 8; ++r) {
    const float il = 1.0f / lrow[r];
    const int s = q0 + r + 8 * hf;
    __bf16* dst = sa + ((size_t)b * S + s) * E + h * DH + n;
    dst[0]  = (__bf16)(o0[r] * il);
    dst[16] = (__bf16)(o1[r] * il);
    dst[32] = (__bf16)(o2[r] * il);
    dst[48] = (__bf16)(o3[r] * il);
  }
}

// ---------------- output projection ----------------
// grid: (B*S/256, E/32), block 256; wave -> one 32x32 tile of [4096 x 1024]
__global__ void __launch_bounds__(256)
oproj_kernel(const __bf16* __restrict__ sa, const __bf16* __restrict__ woT,
             const float* __restrict__ bo, float* __restrict__ out) {
  const int lane = threadIdx.x & 31;
  const int wave = threadIdx.x >> 5;
  const int row0 = blockIdx.x * 256 + wave * 32;   // over B*S
  const int col0 = blockIdx.y * 32;

  const __bf16* A  = sa  + (size_t)row0 * E;
  const __bf16* Bt = woT + (size_t)col0 * E;

  f32x8 c00 = {}, c01 = {}, c10 = {}, c11 = {};
#pragma unroll 4
  for (int k0 = 0; k0 < E; k0 += 32) {
    bf16x16 a0 = load_a_frag(A + k0, E, lane);
    bf16x16 a1 = load_a_frag(A + (size_t)16 * E + k0, E, lane);
    bf16x16 b0 = load_b_frag(Bt + k0, E, lane);
    bf16x16 b1 = load_b_frag(Bt + (size_t)16 * E + k0, E, lane);
    c00 = wmma_bf16(a0, b0, c00);
    c01 = wmma_bf16(a0, b1, c01);
    c10 = wmma_bf16(a1, b0, c10);
    c11 = wmma_bf16(a1, b1, c11);
  }

  const int n = lane & 15, hf = lane >> 4;
  f32x8 cs[4] = {c00, c01, c10, c11};
#pragma unroll
  for (int t = 0; t < 4; ++t) {
    const int col = col0 + (t & 1) * 16 + n;
    const int rbase = row0 + (t >> 1) * 16;
    const float bias = bo[col];
#pragma unroll
    for (int r = 0; r < 8; ++r)
      out[(size_t)(rbase + r + 8 * hf) * E + col] = cs[t][r] + bias;
  }
}

// ---------------- launch ----------------
extern "C" void kernel_launch(void* const* d_in, const int* in_sizes, int n_in,
                              void* d_out, int out_size, void* d_ws, size_t ws_size,
                              hipStream_t stream) {
  (void)in_sizes; (void)n_in; (void)out_size; (void)ws_size;
  const float* x    = (const float*)d_in[0];
  const float* Wkqv = (const float*)d_in[1];
  const float* bkqv = (const float*)d_in[2];
  const float* Wo   = (const float*)d_in[3];
  const float* bo   = (const float*)d_in[4];
  float* out = (float*)d_out;

  char* ws = (char*)d_ws;
  __bf16* xb  = (__bf16*)(ws + OFF_XB);
  __bf16* wkT = (__bf16*)(ws + OFF_WKT);
  __bf16* woT = (__bf16*)(ws + OFF_WOT);
  __bf16* Km  = (__bf16*)(ws + OFF_K);
  __bf16* Qm  = (__bf16*)(ws + OFF_Q);
  __bf16* Vt  = (__bf16*)(ws + OFF_VT);
  __bf16* sa  = (__bf16*)(ws + OFF_SA);

  hipLaunchKernelGGL(cvt_x_kernel, dim3((B * S * E) / 256), dim3(256), 0, stream,
                     x, xb, B * S * E);
  hipLaunchKernelGGL(cvt_wkqv_kernel, dim3((H * F3 * E) / 256), dim3(256), 0, stream,
                     Wkqv, wkT);
  hipLaunchKernelGGL(cvt_wo_kernel, dim3((E * E) / 256), dim3(256), 0, stream,
                     Wo, woT);

  hipLaunchKernelGGL(kqv_kernel, dim3(S / 256, F3 / 32, B * H), dim3(256), 0, stream,
                     xb, wkT, bkqv, Km, Qm, Vt);

  hipLaunchKernelGGL(attn_kernel, dim3(S / 128, 1, B * H), dim3(256), 0, stream,
                     Qm, Km, Vt, sa);

  hipLaunchKernelGGL(oproj_kernel, dim3((B * S) / 256, E / 32, 1), dim3(256), 0, stream,
                     sa, woT, bo, out);
}
